// GGNN_22797686407335
// MI455X (gfx1250) — matrix operand order repaired and checked by reading, not compiled, not run on hardware
//
#include <hip/hip_runtime.h>
#include <hip/hip_bf16.h>
#include <math.h>

// GGNN for MI455X (gfx1250): bf16 WMMA for all dense GEMMs, L2-resident
// message buffer for the edge gather, f32 atomics for segment-sum.

#define N_NODES 50000
#define N_EDGES 800000
#define HDIM    64
#define FDIM    74
#define TTYPES  4
#define GSTEPS  4

typedef __attribute__((ext_vector_type(16))) __bf16 v16bf;
typedef __attribute__((ext_vector_type(8)))  __bf16 v8bf;
typedef __attribute__((ext_vector_type(8)))  float  v8f;

// ---------------------------------------------------------------- WMMA utils

static __device__ __forceinline__ v8f wmma_bf16(v16bf a, v16bf b, v8f c) {
  // D = A(16x32 bf16) * B(32x16 bf16) + C(16x16 f32)
  return __builtin_amdgcn_wmma_f32_16x16x32_bf16(
      /*neg_a=*/false, a, /*neg_b=*/false, b,
      /*c_mod=*/(short)0, c, /*reuse_a=*/false, /*reuse_b=*/false);
}

// A-matrix (16x32, bf16, row-major source, lda elements per row).
// ISA 16-bit A layout: lane<16 -> K 0..7 | 16..23 ; lane>=16 -> K 8..15 | 24..31.
static __device__ __forceinline__ v16bf load_a_bf16(const __bf16* A, int lda,
                                                    int row0, int kbase, int lane) {
  int r = lane & 15, hi = lane >> 4;
  const __bf16* p = A + (size_t)(row0 + r) * lda + kbase + 8 * hi;
  v8bf lo = *(const v8bf*)p;
  v8bf hc = *(const v8bf*)(p + 16);
  v16bf a;
#pragma unroll
  for (int i = 0; i < 8; ++i) { a[i] = lo[i]; a[i + 8] = hc[i]; }
  return a;
}

// Same A layout but from an f32 source (agg), converting in-register.
static __device__ __forceinline__ v16bf load_a_f32(const float* A, int lda,
                                                   int row0, int kbase, int lane) {
  int r = lane & 15, hi = lane >> 4;
  const float* p = A + (size_t)(row0 + r) * lda + kbase + 8 * hi;
  v16bf a;
#pragma unroll
  for (int i = 0; i < 8; ++i) {
    a[i]     = (__bf16)p[i];
    a[i + 8] = (__bf16)p[16 + i];
  }
  return a;
}

// ------------------------------------------------------------- pack B tiles
// Repack an f32 row-major weight W[K][Ncols] into WMMA B-layout bf16 tiles:
// tile(kt,nt) holds 32 lanes x 16 bf16; lane<16 -> K 0..7 | 16..23 ; lane>=16 -> K 8..15 | 24..31.
__global__ void pack_b_kernel(const float* __restrict__ W, __bf16* __restrict__ out,
                              int ncols) {
  int kt = blockIdx.x, nt = blockIdx.y, l = threadIdx.x;
  int hi = l >> 4, c = l & 15;
  __bf16* o = out + (((size_t)kt * gridDim.y + nt) * 32 + l) * 16;
  const float* w = W + (size_t)(kt * 32 + 8 * hi) * ncols + nt * 16 + c;
#pragma unroll
  for (int i = 0; i < 16; ++i) o[i] = (__bf16)w[(size_t)(i < 8 ? i : i + 8) * ncols];
}

// ---------------------------------------------------------------- init pass
// h = h_init = x @ W_in + b_in ; also emit bf16 copies. (0.47 GF, runs once;
// F=74 is not WMMA-friendly, VALU dot is fine: W_in is 19 KB, cache-resident.)
__global__ void init_kernel(const float* __restrict__ x, const float* __restrict__ W_in,
                            const float* __restrict__ b_in, float* __restrict__ h,
                            __bf16* __restrict__ h_bf, __bf16* __restrict__ hinit_bf) {
  int idx = blockIdx.x * blockDim.x + threadIdx.x;
  if (idx >= N_NODES * HDIM) return;
  int n = idx >> 6, o = idx & 63;
  const float* xr = x + (size_t)n * FDIM;
  float acc = b_in[o];
#pragma unroll 2
  for (int f = 0; f < FDIM; ++f) acc = fmaf(xr[f], W_in[(size_t)f * HDIM + o], acc);
  h[idx] = acc;
  h_bf[idx] = (__bf16)acc;
  hinit_bf[idx] = (__bf16)acc;
}

__global__ void zero_kernel(float* __restrict__ p, int n) {
  int idx = blockIdx.x * blockDim.x + threadIdx.x;
  if (idx < n) p[idx] = 0.0f;
}

// ------------------------------------------------------- nt = h @ edge_W[t]
// Output nt[N][T*H] (t-major columns). Grid: (3125 row tiles, 4) x 128 thr;
// wave w handles column tile ct = blockIdx.y*4 + w (16 tiles = T*H/16).
__global__ void __launch_bounds__(128)
nt_kernel(const __bf16* __restrict__ h_bf, const __bf16* __restrict__ pk_edge,
          float* __restrict__ ntb) {
  int lane = threadIdx.x & 31;
  int wave = threadIdx.x >> 5;
  int row0 = blockIdx.x << 4;
  int ct = (blockIdx.y << 2) + wave;       // 0..15
  int t = ct >> 2, ntile = ct & 3;
  v8f acc = {};
#pragma unroll
  for (int kt = 0; kt < 2; ++kt) {
    v16bf a = load_a_bf16(h_bf, HDIM, row0, kt * 32, lane);
    const __bf16* bp = pk_edge + ((((size_t)t * 2 + kt) * 4 + ntile) * 32 + lane) * 16;
    acc = wmma_bf16(a, *(const v16bf*)bp, acc);
  }
  int hi = lane >> 4, c = lane & 15;
  float* out = ntb + (size_t)(row0 + 8 * hi) * (TTYPES * HDIM) + ct * 16 + c;
#pragma unroll
  for (int i = 0; i < 8; ++i) out[(size_t)i * (TTYPES * HDIM)] = acc[i];
}

// ------------------------------------------- edge gather + ReLU + scatter-add
// 16 threads per edge, float4 each. nt is 51 MB -> resident in 192 MB L2.
__global__ void edge_kernel(const float* __restrict__ ntb, const int* __restrict__ etype,
                            const int* __restrict__ src, const int* __restrict__ dst,
                            const float* __restrict__ edge_b, float* __restrict__ agg) {
  long long tid = (long long)blockIdx.x * blockDim.x + threadIdx.x;
  int e = (int)(tid >> 4);
  if (e >= N_EDGES) return;
  int q = ((int)tid & 15) << 2;            // 0,4,...,60
  int t = etype[e];
  int s = src[e];
  int d = dst[e];
  float bb = edge_b[t];
  float4 v = *(const float4*)(ntb + (size_t)s * (TTYPES * HDIM) + t * HDIM + q);
  float m0 = fmaxf(v.x + bb, 0.0f);
  float m1 = fmaxf(v.y + bb, 0.0f);
  float m2 = fmaxf(v.z + bb, 0.0f);
  float m3 = fmaxf(v.w + bb, 0.0f);
  float* ap = agg + (size_t)d * HDIM + q;
  atomicAdd(ap + 0, m0);
  atomicAdd(ap + 1, m1);
  atomicAdd(ap + 2, m2);
  atomicAdd(ap + 3, m3);
}

// -------------------------------------------------------------- fused GRU
// gi = agg@Wih, gh = h@Whh (both K=64 via 2 WMMA k-steps), gates + update
// fused in-register. Wave w owns output columns [16w,16w+16); needs the 3
// gate column blocks -> 6 accumulators (48 VGPRs). In-place h update is safe:
// all h/h_bf reads happen before __syncthreads(), writes after.
__global__ void __launch_bounds__(128)
gru_kernel(const float* __restrict__ agg, float* __restrict__ h,
           __bf16* __restrict__ h_bf,
           const __bf16* __restrict__ pk_wih, const __bf16* __restrict__ pk_whh,
           const float* __restrict__ bih, const float* __restrict__ bhh) {
  int lane = threadIdx.x & 31;
  int wave = threadIdx.x >> 5;             // output col tile 0..3
  int row0 = blockIdx.x << 4;
  int hi = lane >> 4, c = lane & 15;
  int j = wave * 16 + c;                   // output column 0..63

  // --- read everything that this block will later overwrite ---
  v16bf ah0 = load_a_bf16(h_bf, HDIM, row0, 0, lane);
  v16bf ah1 = load_a_bf16(h_bf, HDIM, row0, 32, lane);
  float hold[8];
#pragma unroll
  for (int i = 0; i < 8; ++i) hold[i] = h[(size_t)(row0 + i + 8 * hi) * HDIM + j];
  v16bf aa0 = load_a_f32(agg, HDIM, row0, 0, lane);
  v16bf aa1 = load_a_f32(agg, HDIM, row0, 32, lane);
  __syncthreads();

  v8f gi[3], gh[3];
#pragma unroll
  for (int g = 0; g < 3; ++g) { gi[g] = {}; gh[g] = {}; }
#pragma unroll
  for (int g = 0; g < 3; ++g) {
    int ntile = g * 4 + wave;              // 12 col tiles across 3H=192
    const __bf16* bi0 = pk_wih + (((size_t)0 * 12 + ntile) * 32 + lane) * 16;
    const __bf16* bi1 = pk_wih + (((size_t)1 * 12 + ntile) * 32 + lane) * 16;
    gi[g] = wmma_bf16(aa0, *(const v16bf*)bi0, gi[g]);
    gi[g] = wmma_bf16(aa1, *(const v16bf*)bi1, gi[g]);
    const __bf16* bh0 = pk_whh + (((size_t)0 * 12 + ntile) * 32 + lane) * 16;
    const __bf16* bh1 = pk_whh + (((size_t)1 * 12 + ntile) * 32 + lane) * 16;
    gh[g] = wmma_bf16(ah0, *(const v16bf*)bh0, gh[g]);
    gh[g] = wmma_bf16(ah1, *(const v16bf*)bh1, gh[g]);
  }

  float bi_r = bih[j], bi_z = bih[64 + j], bi_n = bih[128 + j];
  float bh_r = bhh[j], bh_z = bhh[64 + j], bh_n = bhh[128 + j];
#pragma unroll
  for (int i = 0; i < 8; ++i) {
    float rpre = gi[0][i] + bi_r + gh[0][i] + bh_r;
    float zpre = gi[1][i] + bi_z + gh[1][i] + bh_z;
    float npre = gi[2][i] + bi_n;
    float hn   = gh[2][i] + bh_n;
    float r = 1.0f / (1.0f + __expf(-rpre));
    float z = 1.0f / (1.0f + __expf(-zpre));
    float n = tanhf(npre + r * hn);
    float hv = hold[i];
    float hnew = (1.0f - z) * n + z * hv;
    size_t idx = (size_t)(row0 + i + 8 * hi) * HDIM + j;
    h[idx] = hnew;
    h_bf[idx] = (__bf16)hnew;
  }
}

// ---------------------------------------------------------------- MLP head
// tmp = relu([h_init, h] @ W1 + b1) : K=128 -> 4 k-steps (2 from each half)
__global__ void __launch_bounds__(128)
mlp1_kernel(const __bf16* __restrict__ hinit_bf, const __bf16* __restrict__ h_bf,
            const __bf16* __restrict__ pk_w1, const float* __restrict__ b1,
            __bf16* __restrict__ tmp_bf) {
  int lane = threadIdx.x & 31;
  int wave = threadIdx.x >> 5;
  int row0 = blockIdx.x << 4;
  int hi = lane >> 4, c = lane & 15;
  int j = wave * 16 + c;
  v8f acc = {};
#pragma unroll
  for (int kt = 0; kt < 4; ++kt) {
    v16bf a = (kt < 2) ? load_a_bf16(hinit_bf, HDIM, row0, kt * 32, lane)
                       : load_a_bf16(h_bf, HDIM, row0, (kt - 2) * 32, lane);
    const __bf16* bp = pk_w1 + (((size_t)kt * 4 + wave) * 32 + lane) * 16;
    acc = wmma_bf16(a, *(const v16bf*)bp, acc);
  }
  float bb = b1[j];
#pragma unroll
  for (int i = 0; i < 8; ++i) {
    float v = fmaxf(acc[i] + bb, 0.0f);
    tmp_bf[(size_t)(row0 + i + 8 * hi) * HDIM + j] = (__bf16)v;
  }
}

// out = tanh(tmp @ W2 + b2)
__global__ void __launch_bounds__(128)
mlp2_kernel(const __bf16* __restrict__ tmp_bf, const __bf16* __restrict__ pk_w2,
            const float* __restrict__ b2, float* __restrict__ out) {
  int lane = threadIdx.x & 31;
  int wave = threadIdx.x >> 5;
  int row0 = blockIdx.x << 4;
  int hi = lane >> 4, c = lane & 15;
  int j = wave * 16 + c;
  v8f acc = {};
#pragma unroll
  for (int kt = 0; kt < 2; ++kt) {
    v16bf a = load_a_bf16(tmp_bf, HDIM, row0, kt * 32, lane);
    const __bf16* bp = pk_w2 + (((size_t)kt * 4 + wave) * 32 + lane) * 16;
    acc = wmma_bf16(a, *(const v16bf*)bp, acc);
  }
  float bb = b2[j];
#pragma unroll
  for (int i = 0; i < 8; ++i)
    out[(size_t)(row0 + i + 8 * hi) * HDIM + j] = tanhf(acc[i] + bb);
}

// ---------------------------------------------------------------- launcher
static inline size_t align256(size_t x) { return (x + 255) & ~(size_t)255; }

extern "C" void kernel_launch(void* const* d_in, const int* in_sizes, int n_in,
                              void* d_out, int out_size, void* d_ws, size_t ws_size,
                              hipStream_t stream) {
  const float* x       = (const float*)d_in[0];
  const int*   etype   = (const int*)d_in[1];
  const int*   src     = (const int*)d_in[2];
  const int*   dst     = (const int*)d_in[3];
  const float* W_in    = (const float*)d_in[4];
  const float* b_in    = (const float*)d_in[5];
  const float* edge_W  = (const float*)d_in[6];
  const float* edge_b  = (const float*)d_in[7];
  const float* gru_Wih = (const float*)d_in[8];
  const float* gru_Whh = (const float*)d_in[9];
  const float* gru_bih = (const float*)d_in[10];
  const float* gru_bhh = (const float*)d_in[11];
  const float* mlp_W1  = (const float*)d_in[12];
  const float* mlp_b1  = (const float*)d_in[13];
  const float* mlp_W2  = (const float*)d_in[14];
  const float* mlp_b2  = (const float*)d_in[15];
  float* out = (float*)d_out;

  // ---- workspace carve-up (~96 MB) ----
  char* base = (char*)d_ws;
  size_t off = 0;
  float*  h        = (float*)(base + off);  off += align256((size_t)N_NODES * HDIM * 4);
  float*  ntb      = (float*)(base + off);  off += align256((size_t)N_NODES * TTYPES * HDIM * 4);
  float*  agg      = (float*)(base + off);  off += align256((size_t)N_NODES * HDIM * 4);
  __bf16* h_bf     = (__bf16*)(base + off); off += align256((size_t)N_NODES * HDIM * 2);
  __bf16* hinit_bf = (__bf16*)(base + off); off += align256((size_t)N_NODES * HDIM * 2);
  __bf16* tmp_bf   = (__bf16*)(base + off); off += align256((size_t)N_NODES * HDIM * 2);
  __bf16* pk_edge  = (__bf16*)(base + off); off += align256((size_t)TTYPES * 2 * 4 * 32 * 16 * 2);
  __bf16* pk_wih   = (__bf16*)(base + off); off += align256((size_t)2 * 12 * 32 * 16 * 2);
  __bf16* pk_whh   = (__bf16*)(base + off); off += align256((size_t)2 * 12 * 32 * 16 * 2);
  __bf16* pk_w1    = (__bf16*)(base + off); off += align256((size_t)4 * 4 * 32 * 16 * 2);
  __bf16* pk_w2    = (__bf16*)(base + off); off += align256((size_t)2 * 4 * 32 * 16 * 2);
  (void)ws_size; (void)in_sizes; (void)n_in; (void)out_size;

  // ---- pack all weights into WMMA B-layout bf16 (tiny, once per call) ----
  for (int t = 0; t < TTYPES; ++t)
    pack_b_kernel<<<dim3(2, 4), 32, 0, stream>>>(
        edge_W + (size_t)t * HDIM * HDIM, pk_edge + (size_t)t * 2 * 4 * 32 * 16, HDIM);
  pack_b_kernel<<<dim3(2, 12), 32, 0, stream>>>(gru_Wih, pk_wih, 3 * HDIM);
  pack_b_kernel<<<dim3(2, 12), 32, 0, stream>>>(gru_Whh, pk_whh, 3 * HDIM);
  pack_b_kernel<<<dim3(4, 4),  32, 0, stream>>>(mlp_W1, pk_w1, HDIM);
  pack_b_kernel<<<dim3(2, 4),  32, 0, stream>>>(mlp_W2, pk_w2, HDIM);

  // ---- h = h_init = x @ W_in + b_in ----
  init_kernel<<<(N_NODES * HDIM + 255) / 256, 256, 0, stream>>>(
      x, W_in, b_in, h, h_bf, hinit_bf);

  const int row_tiles = N_NODES / 16;  // 3125 exactly

  // ---- 4 propagation steps ----
  for (int s = 0; s < GSTEPS; ++s) {
    zero_kernel<<<(N_NODES * HDIM + 255) / 256, 256, 0, stream>>>(agg, N_NODES * HDIM);
    nt_kernel<<<dim3(row_tiles, 4), 128, 0, stream>>>(h_bf, pk_edge, ntb);
    edge_kernel<<<(N_EDGES * 16 + 255) / 256, 256, 0, stream>>>(
        ntb, etype, src, dst, edge_b, agg);
    gru_kernel<<<row_tiles, 128, 0, stream>>>(
        agg, h, h_bf, pk_wih, pk_whh, gru_bih, gru_bhh);
  }

  // ---- MLP head ----
  mlp1_kernel<<<row_tiles, 128, 0, stream>>>(hinit_bf, h_bf, pk_w1, mlp_b1, tmp_bf);
  mlp2_kernel<<<row_tiles, 128, 0, stream>>>(tmp_bf, pk_w2, mlp_b2, out);
}
